// MultiGraphGalerkinNN_51187420234093
// MI455X (gfx1250) — matrix operand order repaired and checked by reading, not compiled, not run on hardware
//
#include <hip/hip_runtime.h>
#include <math.h>

#define NN 8192
#define NE 65536

typedef __attribute__((ext_vector_type(16))) __bf16 v16bf;
typedef __attribute__((ext_vector_type(8)))  __bf16 v8bf;
typedef __attribute__((ext_vector_type(8)))  float  v8f;

__device__ __forceinline__ float gelu_exact(float x) {
  return 0.5f * x * (1.0f + erff(x * 0.7071067811865476f));
}

__device__ __forceinline__ __bf16 f2bf(float x) {
  unsigned u = __builtin_bit_cast(unsigned, x);
  unsigned r = (u + 0x7FFFu + ((u >> 16) & 1u)) >> 16;
  unsigned short h = (unsigned short)r;
  return __builtin_bit_cast(__bf16, h);
}

// ---- WMMA fragment loaders (bf16, 16x16x32 shape), fully vectorized ----
// A (16x32): lanes 0-15 => M=lane, VGPR0..3: K=0..7,  VGPR4..7: K=16..23
//            lanes 16-31 => M=lane-16, VGPR0..3: K=8..15, VGPR4..7: K=24..31
// per lane: halves [k0+half*8 .. +7] and [k0+16+half*8 .. +7]  (two b128 loads)
__device__ __forceinline__ v16bf load_a_frag(const __bf16* __restrict__ A,
                                             int row, int lda, int k0, int half) {
  const __bf16* p = A + row * lda + k0 + half * 8;
  v8bf lo = *(const v8bf*)(p);
  v8bf hi = *(const v8bf*)(p + 16);
  return __builtin_shufflevector(lo, hi, 0, 1, 2, 3, 4, 5, 6, 7,
                                 8, 9, 10, 11, 12, 13, 14, 15);
}
// B (32x16): lane => n = lane&15, same k order as A: halves [k0+half*8 .. +7] and [k0+16+half*8 .. +7].
// B stored "transposed" (N x K row-major): per-lane two runs of 8 contiguous halves, 16 apart.
__device__ __forceinline__ v16bf load_b_frag(const __bf16* __restrict__ B,
                                             int col, int ldb, int k0, int half) {
  const __bf16* p = B + col * ldb + k0 + half * 8;
  v8bf lo = *(const v8bf*)(p);
  v8bf hi = *(const v8bf*)(p + 16);
  return __builtin_shufflevector(lo, hi, 0, 1, 2, 3, 4, 5, 6, 7,
                                 8, 9, 10, 11, 12, 13, 14, 15);
}

// ---------------- elementwise / prep kernels ----------------
__global__ void k_f32_to_bf16(const float* __restrict__ in, __bf16* __restrict__ out, int n) {
  int i = blockIdx.x * blockDim.x + threadIdx.x;
  if (i < n) out[i] = f2bf(in[i]);
}

// out[c*R + r] = bf16(in[r*C + c])   (R x C  ->  C x R)
__global__ void k_f32_to_bf16_T(const float* __restrict__ in, __bf16* __restrict__ out,
                                int R, int C) {
  int i = blockIdx.x * blockDim.x + threadIdx.x;
  if (i >= R * C) return;
  int r = i / C, c = i % C;
  out[c * R + r] = f2bf(in[r * C + c]);
}

__global__ void k_zero_f32(float* p, int n) {
  int i = blockIdx.x * blockDim.x + threadIdx.x;
  if (i < n) p[i] = 0.0f;
}

// fc0: per node -> af(32), f(32), u(32); also X64bf = bf16 concat(af,u)
__global__ void k_fc0(const float* __restrict__ a,
                      const float* __restrict__ aw, const float* __restrict__ ab,
                      const float* __restrict__ fw, const float* __restrict__ fb,
                      const float* __restrict__ uw, const float* __restrict__ ub,
                      float* __restrict__ AF, float* __restrict__ F, float* __restrict__ U,
                      __bf16* __restrict__ X64bf) {
  int i = blockIdx.x * blockDim.x + threadIdx.x;
  if (i >= NN) return;
  float a0 = a[i * 3], a1 = a[i * 3 + 1], a2 = a[i * 3 + 2];
  float af[32], f[32];
#pragma unroll
  for (int c = 0; c < 32; ++c) af[c] = ab[c] + a0 * aw[c] + a1 * aw[32 + c] + a2 * aw[64 + c];
#pragma unroll
  for (int c = 0; c < 32; ++c) f[c] = fb[c] + fw[c] + a1 * fw[32 + c] + a2 * fw[64 + c];
  for (int c = 0; c < 32; ++c) {
    float s = ub[c];
    for (int j = 0; j < 32; ++j) s += af[j] * uw[j * 32 + c];
    for (int j = 0; j < 32; ++j) s += f[j] * uw[(32 + j) * 32 + c];
    U[i * 32 + c] = s;
    X64bf[i * 64 + 32 + c] = f2bf(s);
  }
#pragma unroll
  for (int c = 0; c < 32; ++c) {
    AF[i * 32 + c] = af[c];
    F[i * 32 + c] = f[c];
    X64bf[i * 64 + c] = f2bf(af[c]);
  }
}

// per-node projections of pf(66) onto K1 top/bottom halves (per-edge gelu gate)
__global__ void k_edgeprep(const float* __restrict__ a,
                           const float* __restrict__ AF, const float* __restrict__ U,
                           const float* __restrict__ k1w, const float* __restrict__ k1b,
                           float* __restrict__ PS, float* __restrict__ PD) {
  int i = blockIdx.x * blockDim.x + threadIdx.x;
  if (i >= NN) return;
  float pf[66];
  for (int c = 0; c < 32; ++c) { pf[c] = AF[i * 32 + c]; pf[32 + c] = U[i * 32 + c]; }
  pf[64] = a[i * 3 + 1];
  pf[65] = a[i * 3 + 2];
  for (int j = 0; j < 8; ++j) {
    float s = 0.0f, d = k1b[j];
    for (int c = 0; c < 66; ++c) {
      s += pf[c] * k1w[c * 8 + j];
      d += pf[c] * k1w[(66 + c) * 8 + j];
    }
    PS[i * 8 + j] = s;
    PD[i * 8 + j] = d;
  }
}

// build WyT (288 x 64 bf16, transposed layout): row col in [0,288), k = input dim
__global__ void k_build_wy(const float* __restrict__ k2w, const float* __restrict__ k2b,
                           __bf16* __restrict__ WyT) {
  int idx = blockIdx.x * blockDim.x + threadIdx.x;
  if (idx >= 288 * 64) return;
  int col = idx / 64, i = idx % 64;
  float v;
  if (col < 256) { int j = col >> 5, o = col & 31; v = k2w[j * 2048 + i * 32 + o]; }
  else           { v = k2b[i * 32 + (col - 256)]; }
  WyT[idx] = f2bf(v);
}

// ---------------- generic WMMA bf16 GEMM: C(f32) = A @ B^T-layout ----------------
// B given as (N x K) row-major. nsplit>1 => split-K with f32 atomic reduction.
__global__ void k_wmma_gemm(const __bf16* __restrict__ A, const __bf16* __restrict__ B,
                            float* __restrict__ C, int Nt, int K,
                            int lda, int ldb, int ldc, int nsplit) {
  int tile = blockIdx.x / nsplit;
  int sp = blockIdx.x % nsplit;
  int tm = tile / Nt, tn = tile % Nt;
  int lane = threadIdx.x & 31;
  int half = lane >> 4, mr = lane & 15;
  int row = tm * 16 + mr, col = tn * 16 + mr;
  int kc = K / nsplit;
  int kbeg = sp * kc, kend = kbeg + kc;
  v8f acc = {};
  for (int k0 = kbeg; k0 < kend; k0 += 32) {
    if (k0 + 32 < kend) __builtin_prefetch(&A[row * lda + k0 + 32], 0, 1);
    v16bf a = load_a_frag(A, row, lda, k0, half);
    v16bf b = load_b_frag(B, col, ldb, k0, half);
    acc = __builtin_amdgcn_wmma_f32_16x16x32_bf16(false, a, false, b, (short)0, acc,
                                                  false, false);
  }
  if (nsplit == 1) {
#pragma unroll
    for (int r = 0; r < 8; ++r) C[(tm * 16 + r + 8 * half) * ldc + col] = acc[r];
  } else {
#pragma unroll
    for (int r = 0; r < 8; ++r) atomicAdd(&C[(tm * 16 + r + 8 * half) * ldc + col], acc[r]);
  }
}

// ---------------- pos0 edge scatter ----------------
__global__ void k_edge(const int* __restrict__ src, const int* __restrict__ dst,
                       const float* __restrict__ PS, const float* __restrict__ PD,
                       const float* __restrict__ Y,
                       float* __restrict__ SUM, float* __restrict__ CNT) {
  int t = blockIdx.x * blockDim.x + threadIdx.x;
  int e = t >> 5, o = t & 31;
  if (e >= NE) return;
  int s = src[e], d = dst[e];
  float h[8];
#pragma unroll
  for (int j = 0; j < 8; ++j) h[j] = gelu_exact(PS[s * 8 + j] + PD[d * 8 + j]);
  float m = Y[s * 288 + 256 + o];
#pragma unroll
  for (int j = 0; j < 8; ++j) m += h[j] * Y[s * 288 + j * 32 + o];
  atomicAdd(&SUM[d * 32 + o], m);
  if (o == 0) atomicAdd(&CNT[d], 1.0f);
}

// pos0 epilogue + pack solver input:
//   XS   (96 x NN) f32 channel-major   (residuals + A-operand source)
//   XSbf (96 x NN) bf16 channel-major  (A operand of XH GEMM)
//   XSbfT(NN x 96) bf16 node-major     (B operand of Ww@XS GEMM)
__global__ void k_node_post(const float* __restrict__ AF, const float* __restrict__ U,
                            const float* __restrict__ F, const float* __restrict__ SUM,
                            const float* __restrict__ CNT, const float* __restrict__ root,
                            float* __restrict__ XS, __bf16* __restrict__ XSbf,
                            __bf16* __restrict__ XSbfT) {
  int i = blockIdx.x * blockDim.x + threadIdx.x;
  if (i >= NN) return;
  float x64[64];
  for (int c = 0; c < 32; ++c) { x64[c] = AF[i * 32 + c]; x64[32 + c] = U[i * 32 + c]; }
  float inv = 1.0f / fmaxf(CNT[i], 1.0f);
  for (int o = 0; o < 32; ++o) {
    float s = SUM[i * 32 + o] * inv;
    for (int c = 0; c < 64; ++c) s += x64[c] * root[c * 32 + o];
    float df = F[i * 32 + o] - s;
    XS[(64 + o) * NN + i] = df;
    XSbf[(64 + o) * NN + i] = f2bf(df);
    XSbfT[i * 96 + 64 + o] = f2bf(df);
  }
#pragma unroll
  for (int c = 0; c < 32; ++c) {
    float v0 = x64[c], v1 = x64[32 + c];
    XS[c * NN + i] = v0;          XSbf[c * NN + i] = f2bf(v0);
    XS[(32 + c) * NN + i] = v1;   XSbf[(32 + c) * NN + i] = f2bf(v1);
    XSbfT[i * 96 + c] = f2bf(v0);
    XSbfT[i * 96 + 32 + c] = f2bf(v1);
  }
}

// spectral mode mix: XH2[o][k] = sum_i XH[i][k] * sp_w[i][o][k]  (96x96x64)
__global__ void k_mode_mix(const float* __restrict__ XH, const float* __restrict__ spw,
                           __bf16* __restrict__ XH2bf) {
  int idx = blockIdx.x * blockDim.x + threadIdx.x;
  if (idx >= 96 * 64) return;
  int o = idx / 64, k = idx % 64;
  float s = 0.0f;
  for (int i = 0; i < 96; ++i) s += XH[i * 64 + k] * spw[i * 96 * 64 + o * 64 + k];
  XH2bf[o * 64 + k] = f2bf(s);
}

// fused: acc = XH2 @ bs^T + Ww @ XS ; XN(node-major bf16) = XS + gelu(acc + w_b)
__global__ void k_solver_mix(const __bf16* __restrict__ XH2bf, const __bf16* __restrict__ bsbf,
                             const __bf16* __restrict__ Wwbf, const __bf16* __restrict__ XSbfT,
                             const float* __restrict__ wbias, const float* __restrict__ XS,
                             __bf16* __restrict__ XNbfT) {
  int tile = blockIdx.x;              // 6 * 512 tiles
  int tm = tile / 512, tn = tile % 512;
  int lane = threadIdx.x & 31;
  int half = lane >> 4, mr = lane & 15;
  int row = tm * 16 + mr, col = tn * 16 + mr;
  v8f acc = {};
#pragma unroll
  for (int k0 = 0; k0 < 64; k0 += 32) {   // x1 = XH2(96x64) @ bases^T ; bsbf is (NN x 64)
    v16bf a = load_a_frag(XH2bf, row, 64, k0, half);
    v16bf b = load_b_frag(bsbf, col, 64, k0, half);
    acc = __builtin_amdgcn_wmma_f32_16x16x32_bf16(false, a, false, b, (short)0, acc,
                                                  false, false);
  }
#pragma unroll
  for (int k0 = 0; k0 < 96; k0 += 32) {   // x2 = Ww(96x96) @ XS ; XSbfT is (NN x 96)
    v16bf a = load_a_frag(Wwbf, row, 96, k0, half);
    v16bf b = load_b_frag(XSbfT, col, 96, k0, half);
    acc = __builtin_amdgcn_wmma_f32_16x16x32_bf16(false, a, false, b, (short)0, acc,
                                                  false, false);
  }
#pragma unroll
  for (int r = 0; r < 8; ++r) {
    int m = tm * 16 + r + 8 * half;
    float g = gelu_exact(acc[r] + wbias[m]);
    float xn = XS[m * NN + col] + g;
    XNbfT[col * 96 + m] = f2bf(xn);
  }
}

// USOL[node][c] = U[node][c] + (fc_w @ XN)[c][node] + fc_b[c]  ; XNbfT is (NN x 96)
__global__ void k_sol_out(const __bf16* __restrict__ fcwbf, const __bf16* __restrict__ XNbfT,
                          const float* __restrict__ fcb, const float* __restrict__ U,
                          float* __restrict__ USOL) {
  int tile = blockIdx.x;              // 2 * 512 tiles
  int tm = tile / 512, tn = tile % 512;
  int lane = threadIdx.x & 31;
  int half = lane >> 4, mr = lane & 15;
  int row = tm * 16 + mr, col = tn * 16 + mr;
  v8f acc = {};
#pragma unroll
  for (int k0 = 0; k0 < 96; k0 += 32) {
    v16bf a = load_a_frag(fcwbf, row, 96, k0, half);
    v16bf b = load_b_frag(XNbfT, col, 96, k0, half);
    acc = __builtin_amdgcn_wmma_f32_16x16x32_bf16(false, a, false, b, (short)0, acc,
                                                  false, false);
  }
#pragma unroll
  for (int r = 0; r < 8; ++r) {
    int m = tm * 16 + r + 8 * half;
    USOL[col * 32 + m] = U[col * 32 + m] + acc[r] + fcb[m];
  }
}

// final MLP: out = gelu(u @ w1 + b1) @ w2 + b2
__global__ void k_fc1(const float* __restrict__ USOL, const float* __restrict__ w1,
                      const float* __restrict__ b1, const float* __restrict__ w2,
                      const float* __restrict__ b2, float* __restrict__ out) {
  int i = blockIdx.x * blockDim.x + threadIdx.x;
  if (i >= NN) return;
  float u[32];
#pragma unroll
  for (int c = 0; c < 32; ++c) u[c] = USOL[i * 32 + c];
  float acc = b2[0];
  for (int j = 0; j < 64; ++j) {
    float s = b1[j];
    for (int c = 0; c < 32; ++c) s += u[c] * w1[c * 64 + j];
    acc += gelu_exact(s) * w2[j];
  }
  out[i] = acc;
}

extern "C" void kernel_launch(void* const* d_in, const int* in_sizes, int n_in,
                              void* d_out, int out_size, void* d_ws, size_t ws_size,
                              hipStream_t stream) {
  (void)in_sizes; (void)n_in; (void)out_size; (void)ws_size;
  const float* a       = (const float*)d_in[0];
  const float* bases   = (const float*)d_in[1];
  const float* wbases  = (const float*)d_in[2];
  const int*   e_pos0  = (const int*)d_in[3];
  const float* fc0_a_w = (const float*)d_in[7];
  const float* fc0_a_b = (const float*)d_in[8];
  const float* fc0_f_w = (const float*)d_in[9];
  const float* fc0_f_b = (const float*)d_in[10];
  const float* fc0_u_w = (const float*)d_in[11];
  const float* fc0_u_b = (const float*)d_in[12];
  const float* p0_k1_w = (const float*)d_in[13];
  const float* p0_k1_b = (const float*)d_in[14];
  const float* p0_k2_w = (const float*)d_in[15];
  const float* p0_k2_b = (const float*)d_in[16];
  const float* p0_root = (const float*)d_in[17];
  const float* s0_sp_w = (const float*)d_in[23];
  const float* s0_w_w  = (const float*)d_in[24];
  const float* s0_w_b  = (const float*)d_in[25];
  const float* s0_fc_w = (const float*)d_in[26];
  const float* s0_fc_b = (const float*)d_in[27];
  const float* fc1_w1  = (const float*)d_in[48];
  const float* fc1_b1  = (const float*)d_in[49];
  const float* fc1_w2  = (const float*)d_in[50];
  const float* fc1_b2  = (const float*)d_in[51];
  float* out = (float*)d_out;

  // workspace layout (all carve-outs 256B aligned)
  char* ws = (char*)d_ws;
  size_t o = 0;
  auto alloc = [&](size_t bytes) { size_t r = o; o = (o + bytes + 255) & ~(size_t)255; return r; };
  float*  AF    = (float*)(ws + alloc((size_t)NN * 32 * 4));
  float*  F     = (float*)(ws + alloc((size_t)NN * 32 * 4));
  float*  U     = (float*)(ws + alloc((size_t)NN * 32 * 4));
  __bf16* X64bf = (__bf16*)(ws + alloc((size_t)NN * 64 * 2));
  float*  PS    = (float*)(ws + alloc((size_t)NN * 8 * 4));
  float*  PD    = (float*)(ws + alloc((size_t)NN * 8 * 4));
  __bf16* WyT   = (__bf16*)(ws + alloc((size_t)288 * 64 * 2));
  float*  Y     = (float*)(ws + alloc((size_t)NN * 288 * 4));
  float*  SUM   = (float*)(ws + alloc((size_t)NN * 32 * 4));
  float*  CNT   = (float*)(ws + alloc((size_t)NN * 4));
  float*  XS    = (float*)(ws + alloc((size_t)96 * NN * 4));
  __bf16* XSbf  = (__bf16*)(ws + alloc((size_t)96 * NN * 2));
  __bf16* XSbfT = (__bf16*)(ws + alloc((size_t)NN * 96 * 2));
  __bf16* wbT   = (__bf16*)(ws + alloc((size_t)64 * NN * 2));   // wbases^T (64 x NN)
  __bf16* bsbf  = (__bf16*)(ws + alloc((size_t)NN * 64 * 2));   // bases (NN x 64)
  float*  XH    = (float*)(ws + alloc((size_t)96 * 64 * 4));
  __bf16* XH2bf = (__bf16*)(ws + alloc((size_t)96 * 64 * 2));
  __bf16* Wwbf  = (__bf16*)(ws + alloc((size_t)96 * 96 * 2));
  __bf16* fcwbf = (__bf16*)(ws + alloc((size_t)32 * 96 * 2));
  __bf16* XNbfT = (__bf16*)(ws + alloc((size_t)NN * 96 * 2));
  float*  USOL  = (float*)(ws + alloc((size_t)NN * 32 * 4));

  const int T = 256;
  // precision conversions / layout transforms
  k_f32_to_bf16_T<<<(NN * 64 + T - 1) / T, T, 0, stream>>>(wbases, wbT, NN, 64);
  k_f32_to_bf16<<<(NN * 64 + T - 1) / T, T, 0, stream>>>(bases, bsbf, NN * 64);
  k_f32_to_bf16<<<(96 * 96 + T - 1) / T, T, 0, stream>>>(s0_w_w, Wwbf, 96 * 96);
  k_f32_to_bf16<<<(32 * 96 + T - 1) / T, T, 0, stream>>>(s0_fc_w, fcwbf, 32 * 96);
  k_build_wy<<<(288 * 64 + T - 1) / T, T, 0, stream>>>(p0_k2_w, p0_k2_b, WyT);

  // fc0 + edge prep
  k_fc0<<<NN / T, T, 0, stream>>>(a, fc0_a_w, fc0_a_b, fc0_f_w, fc0_f_b,
                                  fc0_u_w, fc0_u_b, AF, F, U, X64bf);
  k_edgeprep<<<NN / T, T, 0, stream>>>(a, AF, U, p0_k1_w, p0_k1_b, PS, PD);

  // Y = X64(8192x64) @ Wy(64x288)   [B given as WyT (288x64)]
  k_wmma_gemm<<<(NN / 16) * 18, 32, 0, stream>>>(X64bf, WyT, Y, 18, 64, 64, 64, 288, 1);

  // pos0 graph conv scatter
  k_zero_f32<<<(NN * 33 + T - 1) / T, T, 0, stream>>>(SUM, NN * 33); // SUM + CNT contiguous
  k_edge<<<(NE * 32) / T, T, 0, stream>>>(e_pos0, e_pos0 + NE, PS, PD, Y, SUM, CNT);
  k_node_post<<<NN / T, T, 0, stream>>>(AF, U, F, SUM, CNT, p0_root, XS, XSbf, XSbfT);

  // solver 0
  // XH = XS(96x8192) @ wbases(8192x64)  [B given as wbT (64x8192)], split-K x16
  k_zero_f32<<<(96 * 64 + T - 1) / T, T, 0, stream>>>(XH, 96 * 64);
  k_wmma_gemm<<<6 * 4 * 16, 32, 0, stream>>>(XSbf, wbT, XH, 4, NN, NN, NN, 64, 16);
  k_mode_mix<<<(96 * 64 + T - 1) / T, T, 0, stream>>>(XH, s0_sp_w, XH2bf);
  // XN = XS + gelu(XH2 @ bases^T + Ww @ XS + w_b)   (written node-major bf16)
  k_solver_mix<<<6 * (NN / 16), 32, 0, stream>>>(XH2bf, bsbf, Wwbf, XSbfT, s0_w_b, XS, XNbfT);
  // USOL = U + fc_w @ XN + fc_b
  k_sol_out<<<2 * (NN / 16), 32, 0, stream>>>(fcwbf, XNbfT, s0_fc_b, U, USOL);

  // final MLP
  k_fc1<<<NN / T, T, 0, stream>>>(USOL, fc1_w1, fc1_b1, fc1_w2, fc1_b2, out);
}
